// Attention_68702296867180
// MI455X (gfx1250) — matrix operand order repaired and checked by reading, not compiled, not run on hardware
//
#include <hip/hip_runtime.h>
#include <hip/hip_bf16.h>

// ---- problem constants (hardcoded from reference) ----
constexpr int Bc  = 4;
constexpr int Sc  = 2048;
constexpr int Dc  = 2048;
constexpr int Hc  = 16;
constexpr int DHc = 128;   // D / H
constexpr int Rc  = 16;    // LoRA rank
constexpr float LORA_SCALE = 2.0f;

typedef float  v8f   __attribute__((ext_vector_type(8)));
typedef __bf16 v16bf __attribute__((ext_vector_type(16)));
typedef __attribute__((address_space(3))) __bf16 lds_bf16;

union BF16Frag { uint4 q[2]; v16bf v; };

// A-matrix 16x32 bf16 fragment: lane holds row M=lane%16,
// K in {c..c+7, c+16..c+23} with c = (lane>=16)?8:0  -> loads at p and p+16
__device__ inline v16bf ld_frag_a(const __bf16* p) {
  BF16Frag t;
  t.q[0] = *reinterpret_cast<const uint4*>(p);
  t.q[1] = *reinterpret_cast<const uint4*>(p + 16);
  return t.v;
}
// B-matrix 32x16 bf16 fragment: lane holds col N=lane%16,
// K in {c..c+7, c+16..c+23} with c = (lane>=16)?8:0  -> loads at p and p+16
__device__ inline v16bf ld_frag_b(const __bf16* p) {
  BF16Frag t;
  t.q[0] = *reinterpret_cast<const uint4*>(p);
  t.q[1] = *reinterpret_cast<const uint4*>(p + 16);
  return t.v;
}

__device__ inline v8f wmma_bf16(v16bf a, v16bf b, v8f c) {
  return __builtin_amdgcn_wmma_f32_16x16x32_bf16(
      /*neg_a=*/false, a, /*neg_b=*/false, b,
      /*c_mod=*/(short)0, c, /*reuse_a=*/false, /*reuse_b=*/false);
}

// ---- gfx1250 async copy global -> LDS (ASYNCcnt), 16 bytes per lane ----
__device__ inline void async_cp16(__bf16* lds, const __bf16* g) {
  lds_bf16* l3 = (lds_bf16*)lds;   // addrspace(3) pointer: 32-bit VGPR operand
  asm volatile("global_load_async_to_lds_b128 %0, %1, off"
               :: "v"(l3), "v"(g) : "memory");
}
__device__ inline void wait_async_le8() {
  asm volatile("s_wait_asynccnt 0x8" ::: "memory");
}
__device__ inline void wait_async_le6() {
  asm volatile("s_wait_asynccnt 0x6" ::: "memory");
}
__device__ inline void wait_async_le0() {
  asm volatile("s_wait_asynccnt 0x0" ::: "memory");
}

// ---------------------------------------------------------------- converts
__global__ void f32_to_bf16_kernel(const float* __restrict__ in,
                                   __bf16* __restrict__ out, int n) {
  int i = blockIdx.x * blockDim.x + threadIdx.x;
  if (i < n) out[i] = (__bf16)in[i];
}

// W_eff[n,k] = W[n,k] + LORA_SCALE * sum_r up[n,r]*down[r,k]   (bf16 out)
__global__ void fuse_lora_kernel(const float* __restrict__ w,
                                 const float* __restrict__ down,  // [R,D]
                                 const float* __restrict__ up,    // [D,R]
                                 __bf16* __restrict__ out) {
  int i = blockIdx.x * blockDim.x + threadIdx.x;
  if (i >= Dc * Dc) return;
  int nrow = i >> 11;          // /D
  int k    = i & (Dc - 1);
  float acc = w[i];
  #pragma unroll
  for (int r = 0; r < Rc; ++r)
    acc += LORA_SCALE * up[nrow * Rc + r] * down[r * Dc + k];
  out[i] = (__bf16)acc;
}

// ---------------------------------------------------------------- RoPE
// t layout [B,H,S,DH], pairs are interleaved (even,odd)
__global__ void rope_kernel(__bf16* __restrict__ t,
                            const float* __restrict__ cs,   // [S, DH/2]
                            const float* __restrict__ sn) {
  int i = blockIdx.x * blockDim.x + threadIdx.x;     // B*H*S*(DH/2)
  if (i >= Bc * Hc * Sc * (DHc / 2)) return;
  int p   = i & 63;          // pair index within head dim
  int bhs = i >> 6;
  int s   = bhs & (Sc - 1);
  size_t off = (size_t)bhs * DHc + 2 * p;
  float a  = (float)t[off];
  float b  = (float)t[off + 1];
  float c  = cs[s * (DHc / 2) + p];
  float s_ = sn[s * (DHc / 2) + p];
  t[off]     = (__bf16)(a * c - b * s_);
  t[off + 1] = (__bf16)(a * s_ + b * c);
}

// ---------------------------------------------------------------- GEMM
// C[M,N] = A[M,K] * W[N,K]^T  with M=8192, N=K=2048, all bf16, f32 accum.
// Block = 128 threads (4 waves). Block tile 64(M) x 128(N).
// Double-buffered async-copy pipeline (global_load_async_to_lds_b128 +
// s_wait_asynccnt), fragments read from LDS with ds_load_b128.
// EPI: 1 = scatter to Q/K [B,H,S,DH] bf16, 2 = scatter to V^T [B,H,DH,S] bf16,
//      3 = fp32 row-major (final output), 0 = bf16 row-major.
template <int EPI>
__global__ __launch_bounds__(128) void gemm_wmma_kernel(
    const __bf16* __restrict__ A, const __bf16* __restrict__ W,
    void* __restrict__ outp) {
  constexpr int TK  = 32;
  constexpr int ASZ = 64 * TK;    // bf16 elems per A tile
  constexpr int BSZ = 128 * TK;   // bf16 elems per B tile
  __shared__ __bf16 smem[2 * (ASZ + BSZ)];   // 24 KB

  const int tid  = threadIdx.x;
  const int lane = tid & 31;
  const int wave = tid >> 5;
  const int m0 = blockIdx.x * 64;
  const int n0 = blockIdx.y * 128;
  const int l15 = lane & 15;
  const int hi  = lane >> 4;
  const int ka  = hi ? 8 : 0;    // A-frag K base
  const int kb  = hi ? 8 : 0;    // B-frag K base

  auto issue_stage = [&](int k0, int buf) {
    __bf16* dst = smem + buf * (ASZ + BSZ);
    #pragma unroll
    for (int k = 0; k < 2; ++k) {            // A: 256 x 16B chunks
      int c = tid + 128 * k;
      int row = c >> 2, col = (c & 3) * 8;
      async_cp16(dst + c * 8, A + (size_t)(m0 + row) * Dc + k0 + col);
    }
    #pragma unroll
    for (int k = 0; k < 4; ++k) {            // B: 512 x 16B chunks
      int c = tid + 128 * k;
      int row = c >> 2, col = (c & 3) * 8;
      async_cp16(dst + ASZ + c * 8, W + (size_t)(n0 + row) * Dc + k0 + col);
    }
  };

  v8f z = {};
  v8f c[4][2];
  #pragma unroll
  for (int i = 0; i < 4; ++i)
    #pragma unroll
    for (int j = 0; j < 2; ++j) c[i][j] = z;

  constexpr int NSTEP = Dc / TK;
  issue_stage(0, 0);
  for (int s = 0; s < NSTEP; ++s) {
    const int buf = s & 1;
    if (s + 1 < NSTEP) {
      issue_stage((s + 1) * TK, buf ^ 1);
      wait_async_le6();     // previous stage's 6 ops complete (in-order)
    } else {
      wait_async_le0();
    }
    __syncthreads();

    const __bf16* As = smem + buf * (ASZ + BSZ);
    const __bf16* Bs = As + ASZ;
    v16bf bf0 = ld_frag_b(Bs + (wave * 32 + l15) * TK + kb);
    v16bf bf1 = ld_frag_b(Bs + (wave * 32 + 16 + l15) * TK + kb);
    #pragma unroll
    for (int i = 0; i < 4; ++i) {
      v16bf af = ld_frag_a(As + (16 * i + l15) * TK + ka);
      c[i][0] = wmma_bf16(af, bf0, c[i][0]);
      c[i][1] = wmma_bf16(af, bf1, c[i][1]);
    }
    __syncthreads();   // all waves done reading before buffer is re-filled
  }

  const int hi8 = hi * 8;
  const int nw  = n0 + wave * 32;
  #pragma unroll
  for (int i = 0; i < 4; ++i)
    #pragma unroll
    for (int j = 0; j < 2; ++j)
      #pragma unroll
      for (int r = 0; r < 8; ++r) {
        int m = m0 + 16 * i + r + hi8;
        int n = nw + 16 * j + l15;
        float v = c[i][j][r];
        if constexpr (EPI == 1) {          // Q/K -> [B,H,S,DH]
          int b = m >> 11, s = m & (Sc - 1), h = n >> 7, d = n & (DHc - 1);
          ((__bf16*)outp)[((size_t)(b * Hc + h) * Sc + s) * DHc + d] = (__bf16)v;
        } else if constexpr (EPI == 2) {   // V -> [B,H,DH,S] (transposed)
          int b = m >> 11, s = m & (Sc - 1), h = n >> 7, d = n & (DHc - 1);
          ((__bf16*)outp)[((size_t)(b * Hc + h) * DHc + d) * Sc + s] = (__bf16)v;
        } else if constexpr (EPI == 3) {   // fp32 row-major
          ((float*)outp)[(size_t)m * Dc + n] = v;
        } else {                           // bf16 row-major
          ((__bf16*)outp)[(size_t)m * Dc + n] = (__bf16)v;
        }
      }
}

// ---------------------------------------------------------------- attention
// Online-softmax update for one 16x32 score tile pair; writes P (bf16) to LDS.
template <bool MASKED>
__device__ inline void softmax_update(v8f& s0, v8f& s1, v8f* o,
                                      float* mi, float* li, __bf16* pshw,
                                      int q0, int k0, int l15, int hi8) {
  const float sc = 0.08838834764831845f;   // 1/sqrt(128)
  #pragma unroll
  for (int r = 0; r < 8; ++r) {
    float a  = s0[r] * sc;
    float bb = s1[r] * sc;
    if constexpr (MASKED) {
      const int m = q0 + r + hi8;          // global query row
      const int key0 = k0 + l15;
      if (key0 > m)      a  = -1e30f;
      if (key0 + 16 > m) bb = -1e30f;
    }
    float mx = fmaxf(a, bb);
    mx = fmaxf(mx, __shfl_xor(mx, 1, 32));
    mx = fmaxf(mx, __shfl_xor(mx, 2, 32));
    mx = fmaxf(mx, __shfl_xor(mx, 4, 32));
    mx = fmaxf(mx, __shfl_xor(mx, 8, 32));
    const float mnew = fmaxf(mi[r], mx);
    const float al = __expf(mi[r] - mnew);
    const float p0 = __expf(a - mnew);
    const float p1 = __expf(bb - mnew);
    float rs = p0 + p1;
    rs += __shfl_xor(rs, 1, 32);
    rs += __shfl_xor(rs, 2, 32);
    rs += __shfl_xor(rs, 4, 32);
    rs += __shfl_xor(rs, 8, 32);
    li[r] = li[r] * al + rs;
    mi[r] = mnew;
    #pragma unroll
    for (int t = 0; t < 8; ++t) o[t][r] = o[t][r] * al;
    pshw[(r + hi8) * 32 + l15]      = (__bf16)p0;
    pshw[(r + hi8) * 32 + 16 + l15] = (__bf16)p1;
  }
}

// Block = 128 threads (4 waves) covering 64 q rows; wave w owns rows
// q0 = blk*64 + w*16. K tile (32x128) and V^T tile (128x32) are staged into
// LDS once per block via async copies (double-buffered), shared by all waves.
// Q,K: [B,H,S,DH] bf16 (RoPE applied). Vt: [B,H,DH,S] bf16.
// Output Oa: [B*S, D] bf16 row-major.
__global__ __launch_bounds__(128) void attn_kernel(
    const __bf16* __restrict__ Q, const __bf16* __restrict__ K,
    const __bf16* __restrict__ Vt, __bf16* __restrict__ Oa) {
  constexpr int KTS = 32 * DHc;   // K tile elems (32 keys x 128 dims)
  constexpr int VTS = DHc * 32;   // V tile elems (128 dims x 32 keys)
  __shared__ __bf16 ktile[2][KTS];   // 2 x 8KB
  __shared__ __bf16 vtile[2][VTS];   // 2 x 8KB
  __shared__ __bf16 psh[4][16 * 32]; // per-wave P tile, 4KB

  const int tid  = threadIdx.x;
  const int lane = tid & 31;
  const int wave = tid >> 5;
  const int q0blk = blockIdx.x * 64;
  const int q0 = q0blk + wave * 16;
  const int h  = blockIdx.y;
  const int b  = blockIdx.z;
  const int l15  = lane & 15;
  const int hi   = lane >> 4;
  const int ka   = hi ? 8 : 0;
  const int kb   = hi ? 8 : 0;
  const int hi8  = hi * 8;

  const size_t bh = (size_t)(b * Hc + h);
  const __bf16* Qb = Q  + bh * Sc * DHc;
  const __bf16* Kb = K  + bh * Sc * DHc;
  const __bf16* Vb = Vt + bh * DHc * Sc;   // [DH][S]

  // stage K/V tiles for keys [k0, k0+32) into buffer `buf` (8 ops/thread)
  auto issue_stage = [&](int k0, int buf) {
    #pragma unroll
    for (int k = 0; k < 4; ++k) {            // K: 512 x 16B chunks
      int c = tid + 128 * k;
      int key = c >> 4, col = (c & 15) * 8;
      async_cp16(&ktile[buf][c * 8], Kb + (size_t)(k0 + key) * DHc + col);
    }
    #pragma unroll
    for (int k = 0; k < 4; ++k) {            // V: 512 x 16B chunks
      int c = tid + 128 * k;
      int dim = c >> 2, col = (c & 3) * 8;
      async_cp16(&vtile[buf][c * 8], Vb + (size_t)dim * Sc + k0 + col);
    }
  };

  // Q fragments: 4 chunks of K-dim (DH=128)
  v16bf qf[4];
  #pragma unroll
  for (int cch = 0; cch < 4; ++cch)
    qf[cch] = ld_frag_a(Qb + (size_t)(q0 + l15) * DHc + 32 * cch + ka);

  v8f z = {};
  v8f o[8];
  #pragma unroll
  for (int t = 0; t < 8; ++t) o[t] = z;
  float mi[8], li[8];
  #pragma unroll
  for (int r = 0; r < 8; ++r) { mi[r] = -1e30f; li[r] = 0.0f; }

  const int nblk = (q0blk + 64 + 31) / 32;   // keys < q0blk+64 (causal)

  issue_stage(0, 0);
  for (int jb = 0; jb < nblk; ++jb) {
    const int k0 = jb * 32;
    const int buf = jb & 1;
    if (jb + 1 < nblk) {
      issue_stage((jb + 1) * 32, buf ^ 1);
      wait_async_le8();    // previous stage's 8 ops complete (in-order)
    } else {
      wait_async_le0();
    }
    __syncthreads();

    if (k0 <= q0 + 15) {   // block contributes to this wave's rows
      const __bf16* Ks = ktile[buf];
      const __bf16* Vs = vtile[buf];
      v8f s0 = z, s1 = z;
      #pragma unroll
      for (int cch = 0; cch < 4; ++cch) {
        v16bf kf0 = ld_frag_b(Ks + (l15)      * DHc + 32 * cch + kb);
        v16bf kf1 = ld_frag_b(Ks + (16 + l15) * DHc + 32 * cch + kb);
        s0 = wmma_bf16(qf[cch], kf0, s0);
        s1 = wmma_bf16(qf[cch], kf1, s1);
      }
      if (k0 + 31 <= q0) {  // fully inside causal region: no masking needed
        softmax_update<false>(s0, s1, o, mi, li, psh[wave], q0, k0, l15, hi8);
      } else {              // diagonal block
        softmax_update<true>(s0, s1, o, mi, li, psh[wave], q0, k0, l15, hi8);
      }
      // P (16x32) as A-fragment from LDS (same-wave DS ops are in-order)
      v16bf pf = ld_frag_a(psh[wave] + l15 * 32 + ka);
      #pragma unroll
      for (int t = 0; t < 8; ++t) {
        v16bf vf = ld_frag_b(Vs + (16 * t + l15) * 32 + kb);
        o[t] = wmma_bf16(pf, vf, o[t]);
      }
    }
    __syncthreads();   // all waves done reading before buffer is re-filled
  }

  // normalize + store: Oa[(b*S + q0 + row) * D + h*DH + 16t + l15]
  #pragma unroll
  for (int t = 0; t < 8; ++t)
    #pragma unroll
    for (int r = 0; r < 8; ++r) {
      const int row = r + hi8;
      const float v = o[t][r] / li[r];
      Oa[(size_t)(b * Sc + q0 + row) * Dc + h * DHc + 16 * t + l15] = (__bf16)v;
    }
}

// ---------------------------------------------------------------- launch
extern "C" void kernel_launch(void* const* d_in, const int* in_sizes, int n_in,
                              void* d_out, int out_size, void* d_ws,
                              size_t ws_size, hipStream_t stream) {
  const float* x       = (const float*)d_in[0];
  const float* wq      = (const float*)d_in[1];
  const float* wk      = (const float*)d_in[2];
  const float* wv      = (const float*)d_in[3];
  const float* wo      = (const float*)d_in[4];
  const float* lq_down = (const float*)d_in[5];
  const float* lq_up   = (const float*)d_in[6];
  const float* lk_down = (const float*)d_in[7];
  const float* lk_up   = (const float*)d_in[8];
  const float* lv_down = (const float*)d_in[9];
  const float* lv_up   = (const float*)d_in[10];
  const float* fcos    = (const float*)d_in[11];
  const float* fsin    = (const float*)d_in[12];
  // d_in[13] = mask (unused; causal mask computed in-kernel)

  const size_t M   = (size_t)Bc * Sc;          // 8192
  const size_t xbN = M * Dc;                   // 16,777,216 elems
  const size_t wN  = (size_t)Dc * Dc;          // 4,194,304 elems
  const size_t qkN = (size_t)Bc * Hc * Sc * DHc;

  char* ws = (char*)d_ws;
  __bf16* xb    = (__bf16*)(ws);                           size_t off = xbN * 2;
  __bf16* wqe   = (__bf16*)(ws + off);                     off += wN * 2;
  __bf16* wke   = (__bf16*)(ws + off);                     off += wN * 2;
  __bf16* wve   = (__bf16*)(ws + off);                     off += wN * 2;
  __bf16* wob   = (__bf16*)(ws + off);                     off += wN * 2;
  __bf16* Qb    = (__bf16*)(ws + off);                     off += qkN * 2;
  __bf16* Kb    = (__bf16*)(ws + off);                     off += qkN * 2;
  __bf16* Vt    = (__bf16*)(ws + off);                     off += qkN * 2;
  __bf16* attnb = (__bf16*)(ws + off);                     off += xbN * 2;
  (void)ws_size; (void)n_in; (void)in_sizes; (void)out_size;

  // 1) converts + LoRA folding
  f32_to_bf16_kernel<<<(int)(xbN / 256), 256, 0, stream>>>(x, xb, (int)xbN);
  fuse_lora_kernel<<<(int)(wN / 256), 256, 0, stream>>>(wq, lq_down, lq_up, wqe);
  fuse_lora_kernel<<<(int)(wN / 256), 256, 0, stream>>>(wk, lk_down, lk_up, wke);
  fuse_lora_kernel<<<(int)(wN / 256), 256, 0, stream>>>(wv, lv_down, lv_up, wve);
  f32_to_bf16_kernel<<<(int)(wN / 256), 256, 0, stream>>>(wo, wob, (int)wN);

  // 2) projection GEMMs (WMMA bf16, async-LDS pipelined)
  dim3 gg((unsigned)(M / 64), Dc / 128);
  gemm_wmma_kernel<1><<<gg, 128, 0, stream>>>(xb, wqe, Qb);
  gemm_wmma_kernel<1><<<gg, 128, 0, stream>>>(xb, wke, Kb);
  gemm_wmma_kernel<2><<<gg, 128, 0, stream>>>(xb, wve, Vt);

  // 3) RoPE on Q and K
  const int ropeN = Bc * Hc * Sc * (DHc / 2);
  rope_kernel<<<ropeN / 256, 256, 0, stream>>>(Qb, fcos, fsin);
  rope_kernel<<<ropeN / 256, 256, 0, stream>>>(Kb, fcos, fsin);

  // 4) flash attention (4 waves / 64 q rows per block, async K/V staging)
  dim3 ga(Sc / 64, Hc, Bc);
  attn_kernel<<<ga, 128, 0, stream>>>(Qb, Kb, Vt, attnb);

  // 5) output projection -> fp32 d_out
  gemm_wmma_kernel<3><<<gg, 128, 0, stream>>>(attnb, wob, (float*)d_out);
}
